// AdvancedLSTMTemporal_89309549953786
// MI455X (gfx1250) — compile-verified, matrix operand order checked by reading, not run on hardware
//
#include <hip/hip_runtime.h>

// ---------------- model dims ----------------
#define BB  128   // batch
#define TT  256   // seq len
#define II  64    // input feat
#define HH  256   // hidden
#define NHH 8     // attn heads
#define DKK 32    // head dim
#define MS  128   // memory slots
#define DD  64    // memory dim
#define NCC 2     // classes

typedef __attribute__((ext_vector_type(16))) __bf16 v16bf;
typedef __attribute__((ext_vector_type(8)))  float  v8f;
typedef __attribute__((ext_vector_type(2)))  float  v2f;
typedef __attribute__((ext_vector_type(2)))  __bf16 v2bf;

__device__ __forceinline__ float sigf(float x) { return 1.0f / (1.0f + __expf(-x)); }

// pack two f32 -> two bf16 in one u32 (hardware packed convert)
__device__ __forceinline__ unsigned int packbf2(float a, float b) {
    v2f vv = {a, b};
    v2bf r = __builtin_convertvector(vv, v2bf);
    unsigned int u;
    __builtin_memcpy(&u, &r, 4);
    return u;
}

// =====================================================================
// bf16-WMMA GEMM:  C = act( (A @ B) * alpha + bias + add )
//  A: f32 [M,K] row-major (lda); B: f32 logical [K,N],
//  transB ? stored [N,K] (B[k,n]=Bm[n*ldb+k]) : stored [K,N].
//  batching: per-matrix offset = (bz/nhDiv)*s?o + (bz%nhDiv)*s?i
//
//  Block = 256 threads (8 waves) -> 128x64 tile of C.
//  LDS tiles staged in *fragment-major* layout: each lane's 16 bf16
//  fragment elements are contiguous -> ds_load_b128 x2 per fragment.
//  Out-of-range staging uses address clamping (no exec-mask branches);
//  garbage only reaches output tiles whose stores are guarded anyway.
// =====================================================================
__global__ void __launch_bounds__(256)
gemm_wmma_kernel(const float* __restrict__ A, int lda, long long sAo, long long sAi,
                 const float* __restrict__ Bm, int ldb, long long sBo, long long sBi, int transB,
                 float* __restrict__ C, int ldc, long long sCo, long long sCi,
                 const float* __restrict__ bias,
                 const float* __restrict__ add, int ldadd, long long sDo, long long sDi,
                 int M, int N, int K, int nhDiv, int act, float alpha)
{
    // A: 8 frags * 512 elems = 4096 bf16 = 2048 u32 ; B: 4 frags * 512 = 1024 u32
    __shared__ __attribute__((aligned(32))) unsigned int lsA32[2048];
    __shared__ __attribute__((aligned(32))) unsigned int lsB32[1024];

    const int bz = blockIdx.z;
    const int bo = bz / nhDiv, bi = bz % nhDiv;
    const long long oA = (long long)bo * sAo + (long long)bi * sAi;
    const long long oB = (long long)bo * sBo + (long long)bi * sBi;
    const long long oC = (long long)bo * sCo + (long long)bi * sCi;
    const long long oD = (long long)bo * sDo + (long long)bi * sDi;

    const int tid  = threadIdx.x;
    const int wv   = tid >> 5;          // wave 0..7 -> 16-row slice
    const int lane = tid & 31;
    const int half = lane >> 4;
    const int l16  = lane & 15;
    const int row0 = blockIdx.y * 128;
    const int col0 = blockIdx.x * 64;

    v8f acc0 = {0,0,0,0,0,0,0,0};
    v8f acc1 = acc0, acc2 = acc0, acc3 = acc0;

    const v16bf* pA = reinterpret_cast<const v16bf*>(lsA32);
    const v16bf* pB = reinterpret_cast<const v16bf*>(lsB32);

    for (int k0 = 0; k0 < K; k0 += 32) {
        // ---- stage A tile 128x32: 2048 K-pairs, 8 per thread ----
        #pragma unroll
        for (int pp = 0; pp < 8; ++pp) {
            int p  = tid + pp * 256;          // pair id 0..2047
            int r  = p >> 4;                  // row in tile 0..127
            int c  = (p & 15) << 1;           // even k-col 0..30
            int gr = row0 + r; if (gr > M - 1) gr = M - 1;   // clamp, no branch
            const float* src = A + oA + (long long)gr * lda + (k0 + c);
            v2f vv = *reinterpret_cast<const v2f*>(src);
            // fragment-major destination
            int f     = r >> 4;
            int laneT = (r & 15) + 16 * ((c >> 3) & 1);
            int e     = ((c >> 4) << 3) | (c & 7);
            lsA32[(f * 512 + laneT * 16 + e) >> 1] = packbf2(vv[0], vv[1]);
        }
        // ---- stage B tile 32x64: 1024 K-pairs, 4 per thread ----
        #pragma unroll
        for (int pp = 0; pp < 4; ++pp) {
            int p  = tid + pp * 256;          // pair id 0..1023
            int r  = (p >> 6) << 1;           // even k-row 0..30
            int c  = p & 63;                  // n-col 0..63
            int gk = k0 + r;
            int gn = col0 + c; if (gn > N - 1) gn = N - 1;   // clamp, no branch
            float v0, v1;
            if (transB) {
                const float* src = Bm + oB + (long long)gn * ldb + gk;
                v2f vv = *reinterpret_cast<const v2f*>(src);
                v0 = vv[0]; v1 = vv[1];
            } else {
                v0 = Bm[oB + (long long)gk * ldb + gn];
                v1 = Bm[oB + (long long)(gk + 1) * ldb + gn];
            }
            int nt    = c >> 4;
            int laneT = (c & 15) + 16 * ((r >> 3) & 1);
            int e     = ((r >> 4) << 3) | (r & 7);
            lsB32[(nt * 512 + laneT * 16 + e) >> 1] = packbf2(v0, v1);
        }
        // prefetch next K-slab of A (lowers to global_prefetch_b8)
        if (k0 + 32 < K) {
            int pr = row0 + (tid >> 1); if (pr > M - 1) pr = M - 1;
            __builtin_prefetch(A + oA + (long long)pr * lda + (k0 + 32) + (tid & 1) * 16, 0, 3);
        }
        __syncthreads();

        // fragment loads: one contiguous 32B LDS read each
        v16bf af  = pA[wv * 32 + lane];
        v16bf bf0 = pB[ 0 + lane];
        v16bf bf1 = pB[32 + lane];
        v16bf bf2 = pB[64 + lane];
        v16bf bf3 = pB[96 + lane];

        acc0 = __builtin_amdgcn_wmma_f32_16x16x32_bf16(false, af, false, bf0, (short)0, acc0, false, false);
        acc1 = __builtin_amdgcn_wmma_f32_16x16x32_bf16(false, af, false, bf1, (short)0, acc1, false, false);
        acc2 = __builtin_amdgcn_wmma_f32_16x16x32_bf16(false, af, false, bf2, (short)0, acc2, false, false);
        acc3 = __builtin_amdgcn_wmma_f32_16x16x32_bf16(false, af, false, bf3, (short)0, acc3, false, false);
        __syncthreads();
    }

    v8f accs[4] = {acc0, acc1, acc2, acc3};
    #pragma unroll
    for (int nt = 0; nt < 4; ++nt) {
        int gc = col0 + nt * 16 + l16;
        if (gc >= N) continue;
        float bv = bias ? bias[gc] : 0.0f;
        #pragma unroll
        for (int r = 0; r < 8; ++r) {
            int gr = row0 + wv * 16 + r + 8 * half;
            if (gr >= M) continue;
            float v = accs[nt][r] * alpha + bv;
            if (add) v += add[oD + (long long)gr * ldadd + gc];
            if (act == 1)      v = fmaxf(v, 0.0f);
            else if (act == 2) v = sigf(v);
            C[oC + (long long)gr * ldc + gc] = v;
        }
    }
}

// =====================================================================
// elementwise / reduction kernels
// =====================================================================
__global__ void fill_kernel(float* p, int n) {
    int i = blockIdx.x * blockDim.x + threadIdx.x;
    if (i < n) p[i] = 0.0f;
}

__global__ void copy2d_kernel(const float* __restrict__ a, int lda,
                              float* __restrict__ o, int ldo, int n, int cols) {
    int i = blockIdx.x * blockDim.x + threadIdx.x;
    if (i >= n) return;
    int r = i / cols, c = i % cols;
    o[(long long)r * ldo + c] = a[(long long)r * lda + c];
}

__global__ void mul_kernel(const float* __restrict__ a, const float* __restrict__ b,
                           float* __restrict__ o, int n) {
    int i = blockIdx.x * blockDim.x + threadIdx.x;
    if (i < n) o[i] = a[i] * b[i];
}

__global__ void mul_sig_kernel(const float* __restrict__ a, const float* __restrict__ g,
                               float* __restrict__ o, int n) {
    int i = blockIdx.x * blockDim.x + threadIdx.x;
    if (i < n) o[i] = a[i] * sigf(g[i]);
}

__global__ void avg3_kernel(const float* __restrict__ a, const float* __restrict__ b,
                            const float* __restrict__ c, float* __restrict__ o,
                            long long ldo, int n) {
    int i = blockIdx.x * blockDim.x + threadIdx.x;
    if (i >= n) return;
    int r = i / HH, j = i % HH;
    o[(long long)r * ldo + j] = (a[i] + b[i] + c[i]) * (1.0f / 3.0f);
}

// torch LSTMCell gate order i,f,g,o; z: [B, 4H]; c updated in place
__global__ void lstm_cell_kernel(const float* __restrict__ z, float* __restrict__ c,
                                 float* __restrict__ hraw, int n) {
    int i = blockIdx.x * blockDim.x + threadIdx.x;
    if (i >= n) return;
    int b = i / HH, j = i % HH;
    const float* zr = z + (long long)b * 4 * HH;
    float ig = sigf(zr[j]);
    float fg = sigf(zr[HH + j]);
    float gg = tanhf(zr[2 * HH + j]);
    float og = sigf(zr[3 * HH + j]);
    float cv = fg * c[i] + ig * gg;
    c[i] = cv;
    hraw[i] = og * tanhf(cv);
}

// row softmax (in place): x <- softmax(x*scale/atemp + tbias[(row/biasDiv)%biasMod])
__global__ void __launch_bounds__(128)
softmax_kernel(float* __restrict__ x, int L,
               const float* __restrict__ tb, int biasDiv, int biasMod,
               const float* __restrict__ atemp, float scale) {
    __shared__ float red[128];
    const int row = blockIdx.x;
    float* p = x + (long long)row * L;
    float sc = scale;
    if (atemp) sc /= atemp[0];
    const float bv = tb ? tb[(row / biasDiv) % biasMod] : 0.0f;
    const int tid = threadIdx.x;

    float mx = -3.4e38f;
    for (int j = tid; j < L; j += 128) {
        float v = p[j] * sc + bv;
        p[j] = v;
        mx = fmaxf(mx, v);
    }
    red[tid] = mx; __syncthreads();
    for (int s = 64; s > 0; s >>= 1) { if (tid < s) red[tid] = fmaxf(red[tid], red[tid + s]); __syncthreads(); }
    mx = red[0]; __syncthreads();

    float sum = 0.0f;
    for (int j = tid; j < L; j += 128) {
        float e = __expf(p[j] - mx);
        p[j] = e; sum += e;
    }
    red[tid] = sum; __syncthreads();
    for (int s = 64; s > 0; s >>= 1) { if (tid < s) red[tid] += red[tid + s]; __syncthreads(); }
    float inv = 1.0f / red[0];
    for (int j = tid; j < L; j += 128) p[j] *= inv;
}

__global__ void __launch_bounds__(256)
layernorm_kernel(float* __restrict__ x, const float* __restrict__ w,
                 const float* __restrict__ b) {
    __shared__ float red[256];
    const int row = blockIdx.x, tid = threadIdx.x;
    float* p = x + (long long)row * HH;
    float v = p[tid];
    red[tid] = v; __syncthreads();
    for (int s = 128; s > 0; s >>= 1) { if (tid < s) red[tid] += red[tid + s]; __syncthreads(); }
    float mu = red[0] * (1.0f / HH); __syncthreads();
    float d = v - mu;
    red[tid] = d * d; __syncthreads();
    for (int s = 128; s > 0; s >>= 1) { if (tid < s) red[tid] += red[tid + s]; __syncthreads(); }
    float var = red[0] * (1.0f / HH);
    p[tid] = d * rsqrtf(var + 1e-5f) * w[tid] + b[tid];
}

// logits = o[:, T-1, :] @ clsW + clsb  (N=2, below WMMA granularity)
__global__ void cls_kernel(const float* __restrict__ o, const float* __restrict__ W,
                           const float* __restrict__ b, float* __restrict__ out) {
    int i = threadIdx.x;
    if (i >= BB * NCC) return;
    int bb = i / NCC, c = i % NCC;
    const float* row = o + ((long long)bb * TT + (TT - 1)) * HH;
    float s = b[c];
    for (int j = 0; j < HH; ++j) s += row[j] * W[j * NCC + c];
    out[i] = s;
}

// =====================================================================
// host-side launch helpers
// =====================================================================
static inline void launch_gemm(hipStream_t s,
    const float* A, int lda, long long sAo, long long sAi,
    const float* Bm, int ldb, long long sBo, long long sBi, int transB,
    float* C, int ldc, long long sCo, long long sCi,
    const float* bias,
    const float* add, int ldadd, long long sDo, long long sDi,
    int M, int N, int K, int batches, int nhDiv, int act, float alpha)
{
    dim3 grid((N + 63) / 64, (M + 127) / 128, batches);
    gemm_wmma_kernel<<<grid, dim3(256), 0, s>>>(A, lda, sAo, sAi, Bm, ldb, sBo, sBi, transB,
                                                C, ldc, sCo, sCi, bias, add, ldadd, sDo, sDi,
                                                M, N, K, nhDiv, act, alpha);
}

static inline void gemm0(hipStream_t s, const float* A, int lda,
                         const float* Bm, int ldb, int transB,
                         float* C, int ldc, const float* bias,
                         const float* add, int ldadd,
                         int M, int N, int K, int act)
{
    launch_gemm(s, A, lda, 0, 0, Bm, ldb, 0, 0, transB, C, ldc, 0, 0,
                bias, add, ldadd, 0, 0, M, N, K, 1, 1, act, 1.0f);
}

#define EWGRID(n) dim3(((n) + 255) / 256), dim3(256), 0, stream

// =====================================================================
extern "C" void kernel_launch(void* const* d_in, const int* in_sizes, int n_in,
                              void* d_out, int out_size, void* d_ws, size_t ws_size,
                              hipStream_t stream)
{
    (void)in_sizes; (void)n_in; (void)out_size; (void)ws_size;
    auto F = [&](int i) { return (const float*)d_in[i]; };

    const float* x     = F(0);
    const float* projW = F(1);  const float* projB = F(2);
    const float *Wih[3], *Whh[3], *bih[3], *bhh[3], *tgW1[3], *tgb1[3], *tgW2[3], *tgb2[3];
    for (int l = 0; l < 3; ++l) {
        int b = 3 + 8 * l;
        Wih[l] = F(b);      Whh[l]  = F(b + 1); bih[l]  = F(b + 2); bhh[l]  = F(b + 3);
        tgW1[l] = F(b + 4); tgb1[l] = F(b + 5); tgW2[l] = F(b + 6); tgb2[l] = F(b + 7);
    }
    const float* lgW[2] = {F(27), F(29)};
    const float* lgb[2] = {F(28), F(30)};
    const float* mem = F(31);
    const float* qW  = F(32); const float* qb_ = F(33);
    const float* kW  = F(34); const float* kb_ = F(35);
    const float* vW  = F(36); const float* vb_ = F(37);
    const float* wqW = F(38); const float* wqb = F(39);
    const float* wkW = F(40); const float* wkb = F(41);
    const float* wvW = F(42); const float* wvb = F(43);
    const float* woW = F(44); const float* wob = F(45);
    const float* tbias = F(46); const float* atemp = F(47);
    const float* lnw = F(48); const float* lnb = F(49);
    const float* clsW = F(50); const float* clsb = F(51);

    float* ws = (float*)d_ws;
    long long off = 0;
    auto alloc = [&](long long n) { float* p = ws + off; off += n; return p; };
    const long long BT = (long long)BB * TT;

    float* xp   = alloc(BT * HH);     // input proj; reused as Q after scan
    float* y    = alloc(BT * HH);
    float* Kbf  = alloc(BT * HH);
    float* Vbf  = alloc(BT * HH);
    float* ctx  = alloc(BT * HH);
    float* obuf = alloc(BT * HH);
    const int CHB = 16;               // attention batch chunk
    float* scb  = alloc((long long)CHB * NHH * TT * TT);
    float* qbuf = alloc(BT * DD);
    float* mrd  = alloc(BT * DD);
    float* msc  = alloc(BT * MS);
    float* keys = alloc((long long)MS * DD);
    float* zb   = alloc((long long)BB * 4 * HH);
    float* hc   = alloc((long long)6 * BB * HH);
    float* hs[3] = {hc, hc + (long long)BB * HH, hc + (long long)2 * BB * HH};
    float* cs[3] = {hc + (long long)3 * BB * HH, hc + (long long)4 * BB * HH, hc + (long long)5 * BB * HH};
    float* hraw = alloc((long long)BB * HH);
    float* ubuf = alloc((long long)BB * (HH / 2));
    float* glin = alloc((long long)BB * HH);
    float* ccb  = alloc((long long)BB * 2 * HH);
    float* gate = alloc((long long)BB * HH);
    float* inp  = alloc((long long)BB * HH);
    float* Qbf  = xp;   // xp is dead after the scan

    // init recurrent state
    fill_kernel<<<EWGRID(6 * BB * HH)>>>(hc, 6 * BB * HH);

    // input projection: xp = x @ projW + projB     [BT,64]x[64,256]
    gemm0(stream, x, II, projW, HH, 0, xp, HH, projB, nullptr, 0, (int)BT, HH, II, 0);

    // ---------------- hierarchical LSTM scan (sequential in T) ----------------
    for (int t = 0; t < TT; ++t) {
        const float* xin = xp + (long long)t * HH;      // [B,H] slice, lda = T*H
        for (int l = 0; l < 3; ++l) {
            const float* lin; int linLd;
            if (l == 0) { lin = xin; linLd = TT * HH; }
            else {
                // level gate: g = sigmoid([out_prev, h_l] @ lgW + lgb); inp = g * out_prev
                copy2d_kernel<<<EWGRID(BB * HH)>>>(hs[l - 1], HH, ccb, 2 * HH, BB * HH, HH);
                copy2d_kernel<<<EWGRID(BB * HH)>>>(hs[l], HH, ccb + HH, 2 * HH, BB * HH, HH);
                gemm0(stream, ccb, 2 * HH, lgW[l - 1], HH, 0, gate, HH, lgb[l - 1],
                      nullptr, 0, BB, HH, 2 * HH, 2);
                mul_kernel<<<EWGRID(BB * HH)>>>(gate, hs[l - 1], inp, BB * HH);
                lin = inp; linLd = HH;
            }
            // z = lin @ Wih + bih ; z += h @ Whh + bhh
            gemm0(stream, lin, linLd, Wih[l], 4 * HH, 0, zb, 4 * HH, bih[l], nullptr, 0, BB, 4 * HH, HH, 0);
            gemm0(stream, hs[l], HH, Whh[l], 4 * HH, 0, zb, 4 * HH, bhh[l], zb, 4 * HH, BB, 4 * HH, HH, 0);
            lstm_cell_kernel<<<EWGRID(BB * HH)>>>(zb, cs[l], hraw, BB * HH);
            // temporal gate: h = hraw * sigmoid(relu(hraw@tgW1+b1)@tgW2+b2)
            gemm0(stream, hraw, HH, tgW1[l], HH / 2, 0, ubuf, HH / 2, tgb1[l], nullptr, 0, BB, HH / 2, HH, 1);
            gemm0(stream, ubuf, HH / 2, tgW2[l], HH, 0, glin, HH, tgb2[l], nullptr, 0, BB, HH, HH / 2, 0);
            mul_sig_kernel<<<EWGRID(BB * HH)>>>(hraw, glin, hs[l], BB * HH);
        }
        avg3_kernel<<<EWGRID(BB * HH)>>>(hs[0], hs[1], hs[2], y + (long long)t * HH,
                                         (long long)TT * HH, BB * HH);
    }

    // ---------------- memory-augmented read ----------------
    gemm0(stream, y, HH, qW, DD, 0, qbuf, DD, qb_, nullptr, 0, (int)BT, DD, HH, 0);
    gemm0(stream, mem, DD, kW, DD, 0, keys, DD, kb_, nullptr, 0, MS, DD, DD, 0);
    gemm0(stream, qbuf, DD, keys, DD, 1, msc, MS, nullptr, nullptr, 0, (int)BT, MS, DD, 0);
    softmax_kernel<<<(int)BT, 128, 0, stream>>>(msc, MS, nullptr, 1, 1, nullptr, 0.125f); // 1/sqrt(64)
    gemm0(stream, msc, MS, mem, DD, 0, mrd, DD, nullptr, nullptr, 0, (int)BT, DD, MS, 0);
    gemm0(stream, mrd, DD, vW, HH, 0, y, HH, vb_, y, HH, (int)BT, HH, DD, 0); // y += read@vW+vb

    // ---------------- multi-head self-attention ----------------
    gemm0(stream, y, HH, wqW, HH, 0, Qbf, HH, wqb, nullptr, 0, (int)BT, HH, HH, 0);
    gemm0(stream, y, HH, wkW, HH, 0, Kbf, HH, wkb, nullptr, 0, (int)BT, HH, HH, 0);
    gemm0(stream, y, HH, wvW, HH, 0, Vbf, HH, wvb, nullptr, 0, (int)BT, HH, HH, 0);

    const float iscDK = 0.17677669529663687f;   // 1/sqrt(32)
    for (int b0 = 0; b0 < BB; b0 += CHB) {
        const long long bOff = (long long)b0 * TT * HH;
        // scores[cb,h] = Q_bh @ K_bh^T   (M=T, N=T, K=dk)
        launch_gemm(stream,
            Qbf + bOff, HH, (long long)TT * HH, DKK,
            Kbf + bOff, HH, (long long)TT * HH, DKK, 1,
            scb, TT, (long long)NHH * TT * TT, (long long)TT * TT,
            nullptr, nullptr, 0, 0, 0,
            TT, TT, DKK, CHB * NHH, NHH, 0, 1.0f);
        // softmax(s/sqrt(dk)/atemp + tbias[h])
        softmax_kernel<<<CHB * NHH * TT, 128, 0, stream>>>(scb, TT, tbias, TT, NHH, atemp, iscDK);
        // ctx_bh = aw @ V_bh   (M=T, N=dk, K=T)
        launch_gemm(stream,
            scb, TT, (long long)NHH * TT * TT, (long long)TT * TT,
            Vbf + bOff, HH, (long long)TT * HH, DKK, 0,
            ctx + bOff, HH, (long long)TT * HH, DKK,
            nullptr, nullptr, 0, 0, 0,
            TT, DKK, TT, CHB * NHH, NHH, 0, 1.0f);
    }

    // output proj + residual, layernorm, classifier
    gemm0(stream, ctx, HH, woW, HH, 0, obuf, HH, wob, y, HH, (int)BT, HH, HH, 0);
    layernorm_kernel<<<(int)BT, 256, 0, stream>>>(obuf, lnw, lnb);
    cls_kernel<<<1, 256, 0, stream>>>(obuf, clsW, clsb, (float*)d_out);
}
